// RelativeBias_773094113358
// MI455X (gfx1250) — compile-verified, matrix operand order checked by reading, not run on hardware
//
#include <hip/hip_runtime.h>
#include <hip/hip_bf16.h>
#include <math.h>

// CDNA5 / gfx1250, wave32. WMMA f16->f32, 16x16x32.
typedef __attribute__((ext_vector_type(16))) _Float16 v16h;
typedef __attribute__((ext_vector_type(8)))  float    v8f;
typedef __attribute__((ext_vector_type(4)))  float    v4f;

#define RB_EPS 1e-6f
#define RB_N   512
#define RB_B   8
#define RB_H   64
#define RB_NH  16

// ---------------------------------------------------------------------------
// Kernel 1: G[b,n,h] = P[b,n,:] @ W1[:,h]
//   P = (x, y, t, log(dx+eps), log(dy+eps), log(dt+eps))
//   metadata columns: x=0, y=1, dx=2, dy=3, t=4, dt=5
// ---------------------------------------------------------------------------
__global__ void rb_g_kernel(const float* __restrict__ meta,
                            const float* __restrict__ W1,
                            float* __restrict__ G) {
    const int row = blockIdx.x;       // b*512 + n, 0..4095
    const int h   = threadIdx.x;      // 0..63
    const float* m = meta + (size_t)row * 6;
    float p0 = m[0];
    float p1 = m[1];
    float p2 = m[4];
    float p3 = __logf(m[2] + RB_EPS);
    float p4 = __logf(m[3] + RB_EPS);
    float p5 = __logf(m[5] + RB_EPS);
    float acc = p0 * W1[0 * RB_H + h]
              + p1 * W1[1 * RB_H + h]
              + p2 * W1[2 * RB_H + h]
              + p3 * W1[3 * RB_H + h]
              + p4 * W1[4 * RB_H + h]
              + p5 * W1[5 * RB_H + h];
    G[(size_t)row * RB_H + h] = acc;
}

// ---------------------------------------------------------------------------
// Kernel 2: out[b,k,i,j] = relu(G[i]-G[j]+b1) @ W2 + b2
// One wave per (b, i, 16-wide j-tile): D(16j x 16k) via two
// v_wmma_f32_16x16x32_f16 (K = 64 split into 2x32).
//
// Per-lane K mapping for 16-bit A/B operands (ISA 7.12.2):
//   half = lane>>4 ; elements e=0..7  -> K = 8*half + e
//                    elements e=8..15 -> K = 16 + 8*half + (e-8)
// A row (M) = lane&15 ; B col (N) = lane&15.
// ---------------------------------------------------------------------------
__global__ void rb_bias_kernel(const float* __restrict__ G,
                               const float* __restrict__ b1,
                               const float* __restrict__ W2,
                               const float* __restrict__ b2,
                               float* __restrict__ out) {
    const int jt   = blockIdx.x;          // 0..31  (j tile)
    const int b    = blockIdx.y;          // 0..7
    const int iz   = blockIdx.z;          // 0..7   (i chunk of 64)
    const int lane = threadIdx.x & 31;
    const int wave = threadIdx.x >> 5;    // 0..7
    const int half = lane >> 4;
    const int n16  = lane & 15;

    const int k0 = 8 * half;              // e=0..7  -> K = k0+e
    const int k1 = 16 + 8 * half;         // e=8..15 -> K = k1+(e-8)

    // B matrices: W2 (64x16) -> f16, column n16, lane-specific K slices.
    v16h bm0, bm1;
#pragma unroll
    for (int e = 0; e < 8; ++e) {
        bm0[e]     = (_Float16)W2[(k0 + e) * RB_NH + n16];
        bm0[e + 8] = (_Float16)W2[(k1 + e) * RB_NH + n16];
        bm1[e]     = (_Float16)W2[(k0 + e + 32) * RB_NH + n16];
        bm1[e + 8] = (_Float16)W2[(k1 + e + 32) * RB_NH + n16];
    }

    // Hoisted j-dependent part: r = b1[h] - G[b, j_row, h]
    const int jrow = jt * 16 + n16;
    const float* __restrict__ gj = G + ((size_t)b * RB_N + jrow) * RB_H;
    float r0[16], r1[16];
#pragma unroll
    for (int e = 0; e < 8; ++e) {
        r0[e]     = b1[k0 + e]      - gj[k0 + e];
        r0[e + 8] = b1[k1 + e]      - gj[k1 + e];
        r1[e]     = b1[k0 + e + 32] - gj[k0 + e + 32];
        r1[e + 8] = b1[k1 + e + 32] - gj[k1 + e + 32];
    }

    const float cinit = b2[n16];

#pragma unroll 1
    for (int s = 0; s < 8; ++s) {
        const int i = iz * 64 + s * 8 + wave;
        const float* __restrict__ gi = G + ((size_t)b * RB_N + i) * RB_H;

        v16h a0, a1;
#pragma unroll
        for (int e = 0; e < 8; ++e) {
            float x0 = gi[k0 + e]      + r0[e];
            float x1 = gi[k1 + e]      + r0[e + 8];
            float x2 = gi[k0 + e + 32] + r1[e];
            float x3 = gi[k1 + e + 32] + r1[e + 8];
            a0[e]     = (_Float16)fmaxf(x0, 0.0f);
            a0[e + 8] = (_Float16)fmaxf(x1, 0.0f);
            a1[e]     = (_Float16)fmaxf(x2, 0.0f);
            a1[e + 8] = (_Float16)fmaxf(x3, 0.0f);
        }

        v8f c;
#pragma unroll
        for (int r = 0; r < 8; ++r) c[r] = cinit;

        // D = A1*B1 + (A0*B0 + C)
        c = __builtin_amdgcn_wmma_f32_16x16x32_f16(
                false, a0, false, bm0, (short)0, c, false, false);
        c = __builtin_amdgcn_wmma_f32_16x16x32_f16(
                false, a1, false, bm1, (short)0, c, false, false);

        // Lane holds D[M = r + 8*half][N = n16] for r=0..7 -> 8 consecutive j.
        float* __restrict__ p =
            out + ((((size_t)b * RB_NH + n16) * RB_N + i) * RB_N
                   + (size_t)jt * 16 + 8 * half);
        v4f lo = { c[0], c[1], c[2], c[3] };
        v4f hi = { c[4], c[5], c[6], c[7] };
        *(v4f*)(p)     = lo;
        *(v4f*)(p + 4) = hi;
    }
}

// ---------------------------------------------------------------------------
extern "C" void kernel_launch(void* const* d_in, const int* in_sizes, int n_in,
                              void* d_out, int out_size, void* d_ws, size_t ws_size,
                              hipStream_t stream) {
    const float* meta = (const float*)d_in[0];   // (8,512,6)
    const float* W1   = (const float*)d_in[1];   // (6,64)
    const float* b1   = (const float*)d_in[2];   // (64,)
    const float* W2   = (const float*)d_in[3];   // (64,16)
    const float* b2   = (const float*)d_in[4];   // (16,)
    float*       outp = (float*)d_out;           // (8,16,512,512)
    float*       G    = (float*)d_ws;            // 8*512*64 f32 = 1 MB

    // Stage 1: first layer collapsed to per-token precompute (1 MB, L2-resident).
    rb_g_kernel<<<RB_B * RB_N, RB_H, 0, stream>>>(meta, W1, G);

    // Stage 2: pairwise relu + 64x16 GEMM on WMMA, streaming stores.
    dim3 grid(RB_N / 16, RB_B, 8);   // (32 j-tiles, 8 batch, 8 i-chunks)
    rb_bias_kernel<<<grid, 256, 0, stream>>>(G, b1, W2, b2, outp);
}
